// TinyGPT_NTP_67147518705861
// MI455X (gfx1250) — matrix operand order repaired and checked by reading, not compiled, not run on hardware
//
#include <hip/hip_runtime.h>
#include <math.h>

// ---------------------------------------------------------------------------
// TinyGPT forward for MI455X (gfx1250, wave32, WMMA + Tensor Data Mover).
// GEMMs: v_wmma_f32_16x16x32_bf16, 128x64 block tile, 32x32 per-wave tile,
//        TDM (tensor_load_to_lds) double-buffered global->LDS staging.
// Attention: flash-style streaming softmax on WMMA, K/V tiles streamed by TDM
//            with the same double-buffered TENSORcnt pipeline.
// Residual stream fp32; weights converted fp32->bf16 once per launch.
// NOTE: no pointer arrays into LDS (ld.lld rejects addrspacecast in static
//       initializers) -- double buffers are selected by integer arithmetic.
// ---------------------------------------------------------------------------

typedef __bf16 bf16_t;
typedef __attribute__((ext_vector_type(16))) __bf16 v16bf;
typedef __attribute__((ext_vector_type(8)))  float  v8f;
typedef __attribute__((ext_vector_type(4)))  unsigned int v4u;
typedef __attribute__((ext_vector_type(8)))  int v8i_t;
typedef __attribute__((ext_vector_type(4)))  int v4i_t;

#define DMODEL  512
#define NHEADS  8
#define DHEAD   64
#define DFF     2048
#define SEQ     2048
#define BATCH   2
#define NLAYERS 6
#define NVOCAB  256
#define NROWS   (BATCH * SEQ)   // 4096 tokens

__device__ __forceinline__ v8f wmma_bf16(v16bf a, v16bf b, v8f c) {
  return __builtin_amdgcn_wmma_f32_16x16x32_bf16(false, a, false, b,
                                                 (short)0, c, false, false);
}

__device__ __forceinline__ bf16_t f2bf(float f) { return (bf16_t)f; }

// ---------------------------------------------------------------------------
// TDM: DMA one 2-D tile (tile_rows x tile_cols bf16) from a row-major
// [rows_total x row_len] bf16 tensor (global) into LDS (packed, row-major).
// D# per CDNA5 ISA ch.8. Issued once per wave (EXEC-ignored); TENSORcnt.
// ---------------------------------------------------------------------------
__device__ __forceinline__ void tdm_load_tile(unsigned lds_byte_off,
                                              const void* gptr,
                                              unsigned row_len,
                                              unsigned rows_total,
                                              unsigned tile_cols,
                                              unsigned tile_rows) {
  unsigned long long ga = (unsigned long long)(uintptr_t)gptr;
  v4u g0;
  g0.x = 1u;                                            // count=1 (valid D#)
  g0.y = lds_byte_off;                                  // LDS dest (bytes)
  g0.z = (unsigned)(ga & 0xffffffffull);                // global_addr[31:0]
  g0.w = (unsigned)((ga >> 32) & 0x01ffffffull)         // global_addr[56:32]
         | (2u << 30);                                  // type=2 ("image")
  v8i_t g1;
  g1[0] = (int)0x00010000u;                             // data_size=1 -> 2B
  g1[1] = (int)((row_len & 0xffffu) << 16);             // tensor_dim0[15:0]
  g1[2] = (int)((row_len >> 16) | ((rows_total & 0xffffu) << 16));
  g1[3] = (int)((rows_total >> 16) | ((tile_cols & 0xffffu) << 16));
  g1[4] = (int)(tile_rows & 0xffffu);                   // tile_dim1
  g1[5] = (int)row_len;                                 // tensor_dim0_stride
  g1[6] = 0;
  g1[7] = 0;
  v4i_t z4 = {0, 0, 0, 0};
#if defined(__clang_major__) && (__clang_major__ >= 23)
  v8i_t z8 = {0, 0, 0, 0, 0, 0, 0, 0};
  __builtin_amdgcn_tensor_load_to_lds(g0, g1, z4, z4, z8, 0);
#else
  __builtin_amdgcn_tensor_load_to_lds(g0, g1, z4, z4, 0);
#endif
}

// low 32 bits of a flat LDS pointer = byte offset within wave LDS aperture
__device__ __forceinline__ unsigned lds_off(const void* p) {
  return (unsigned)(uintptr_t)p;
}

// ---------------------------------------------------------------------------
// fp32 -> bf16 bulk convert (weights)
// ---------------------------------------------------------------------------
__global__ __launch_bounds__(256)
void cvt_f32_bf16_kernel(const float* __restrict__ s, bf16_t* __restrict__ d,
                         size_t n) {
  size_t i = (size_t)blockIdx.x * blockDim.x + threadIdx.x;
  size_t stride = (size_t)gridDim.x * blockDim.x;
  for (; i < n; i += stride) d[i] = f2bf(s[i]);
}

// ---------------------------------------------------------------------------
// token + positional embedding gather
// ---------------------------------------------------------------------------
__global__ __launch_bounds__(128)
void embed_kernel(const int* __restrict__ x, const float* __restrict__ tok,
                  const float* __restrict__ pos, float* __restrict__ h) {
  int row = blockIdx.x;              // b*SEQ + t
  int t   = row & (SEQ - 1);
  int id  = x[row];
  const float* tr = tok + (size_t)id * DMODEL;
  const float* pr = pos + (size_t)t * DMODEL;
  float* hr = h + (size_t)row * DMODEL;
  for (int c = threadIdx.x; c < DMODEL; c += blockDim.x)
    hr[c] = tr[c] + pr[c];
}

// ---------------------------------------------------------------------------
// LayerNorm (fp32 in, bf16 out). One block per row.
// ---------------------------------------------------------------------------
__global__ __launch_bounds__(128)
void layernorm_kernel(const float* __restrict__ x, const float* __restrict__ w,
                      const float* __restrict__ b, bf16_t* __restrict__ out) {
  int row = blockIdx.x;
  const float* xr = x + (size_t)row * DMODEL;
  float v[4], s = 0.f, ss = 0.f;
#pragma unroll
  for (int i = 0; i < 4; i++) {
    float t = xr[threadIdx.x + i * 128];
    v[i] = t; s += t; ss += t * t;
  }
#pragma unroll
  for (int m = 16; m >= 1; m >>= 1) { s += __shfl_xor(s, m); ss += __shfl_xor(ss, m); }
  __shared__ float red[2][4];
  int wave = threadIdx.x >> 5;
  if ((threadIdx.x & 31) == 0) { red[0][wave] = s; red[1][wave] = ss; }
  __syncthreads();
  s  = red[0][0] + red[0][1] + red[0][2] + red[0][3];
  ss = red[1][0] + red[1][1] + red[1][2] + red[1][3];
  float mu   = s * (1.0f / DMODEL);
  float var  = ss * (1.0f / DMODEL) - mu * mu;
  float rstd = rsqrtf(var + 1e-5f);
  bf16_t* orow = out + (size_t)row * DMODEL;
#pragma unroll
  for (int i = 0; i < 4; i++) {
    int c = threadIdx.x + i * 128;
    orow[c] = f2bf((v[i] - mu) * rstd * w[c] + b[c]);
  }
}

// ---------------------------------------------------------------------------
// WMMA GEMM:  C[M,N] = A[M,K](bf16) x W[N,K]^T(bf16), fp32 accumulate.
// Block tile 128x64, 8 waves, per-wave 32x32 (4 WMMA / 32-wide K step).
// TDM double buffering: wave 0 issues step s+1 tiles into the idle buffer,
// waits TENSORcnt<=2 (=> step-s tiles landed), workgroup barriers, computes.
// LDS layout (bf16 elems): [A0: 128*32][A1: 128*32][W0: 64*32][W1: 64*32]
// MODE: 0 fp32 | 1 fp32+residual | 2 bf16 | 3 GELU->bf16
// ---------------------------------------------------------------------------
#define GEMM_A_ELEMS (128 * 32)
#define GEMM_W_ELEMS (64 * 32)

template <int MODE>
__global__ __launch_bounds__(256)
void gemm_bf16_kernel(const bf16_t* __restrict__ A, const bf16_t* __restrict__ W,
                      float* Cf, bf16_t* Cb, const float* resid,
                      int M, int N, int K) {
  extern __shared__ bf16_t smem[];

  const int bm = blockIdx.y * 128;
  const int bn = blockIdx.x * 64;
  const int tid  = threadIdx.x;
  const int wave = tid >> 5;
  const int lane = tid & 31;
  const int hi   = lane >> 4;
  const int ln   = lane & 15;
  const int waveM = wave >> 1;            // 0..3 -> 32-row strip
  const int waveN = wave & 1;             // 0..1 -> 32-col strip

  const int nsteps = K >> 5;

  const unsigned baseA = lds_off(smem);                        // bytes
  const unsigned baseW = baseA + 2u * GEMM_A_ELEMS * 2u;       // bytes

  if (wave == 0) {
    tdm_load_tile(baseA, A + (size_t)bm * K, K, M, 32, 128);
    tdm_load_tile(baseW, W + (size_t)bn * K, K, N, 32, 64);
  }

  v8f acc00 = {}, acc01 = {}, acc10 = {}, acc11 = {};
  const int kb = hi * 8;
  const int r0 = waveM * 32 + ln;
  const int c0 = waveN * 32 + ln;

  for (int s = 0; s < nsteps; s++) {
    if (wave == 0) {
      if (s + 1 < nsteps) {
        unsigned nb = (unsigned)(s + 1) & 1u;
        tdm_load_tile(baseA + nb * (GEMM_A_ELEMS * 2u),
                      A + (size_t)bm * K + (s + 1) * 32, K, M, 32, 128);
        tdm_load_tile(baseW + nb * (GEMM_W_ELEMS * 2u),
                      W + (size_t)bn * K + (s + 1) * 32, K, N, 32, 64);
        __builtin_amdgcn_s_wait_tensorcnt(2);   // step-s tiles complete
      } else {
        __builtin_amdgcn_s_wait_tensorcnt(0);
      }
    }
    __syncthreads();

    const bf16_t* Ab = smem + (s & 1) * GEMM_A_ELEMS;
    const bf16_t* Wb = smem + 2 * GEMM_A_ELEMS + (s & 1) * GEMM_W_ELEMS;

    v16bf a0, a1, b0, b1;
#pragma unroll
    for (int i = 0; i < 8; i++) {
      a0[i]     = Ab[r0 * 32 + kb + i];
      a0[8 + i] = Ab[r0 * 32 + kb + 16 + i];
      a1[i]     = Ab[(r0 + 16) * 32 + kb + i];
      a1[8 + i] = Ab[(r0 + 16) * 32 + kb + 16 + i];
    }
#pragma unroll
    for (int i = 0; i < 16; i++) {
      b0[i] = Wb[c0 * 32 + (i & 8) * 2 + kb + (i & 7)];
      b1[i] = Wb[(c0 + 16) * 32 + (i & 8) * 2 + kb + (i & 7)];
    }
    acc00 = wmma_bf16(a0, b0, acc00);
    acc01 = wmma_bf16(a0, b1, acc01);
    acc10 = wmma_bf16(a1, b0, acc10);
    acc11 = wmma_bf16(a1, b1, acc11);
    __syncthreads();
  }

  const int n0 = bn + waveN * 32 + ln;
  const int n1 = n0 + 16;
#pragma unroll
  for (int j = 0; j < 8; j++) {
    int m0 = bm + waveM * 32 + j + 8 * hi;
    int m1 = m0 + 16;
    float v00 = acc00[j], v01 = acc01[j], v10 = acc10[j], v11 = acc11[j];
    if (MODE == 0) {
      Cf[(size_t)m0 * N + n0] = v00; Cf[(size_t)m0 * N + n1] = v01;
      Cf[(size_t)m1 * N + n0] = v10; Cf[(size_t)m1 * N + n1] = v11;
    } else if (MODE == 1) {
      Cf[(size_t)m0 * N + n0] = resid[(size_t)m0 * N + n0] + v00;
      Cf[(size_t)m0 * N + n1] = resid[(size_t)m0 * N + n1] + v01;
      Cf[(size_t)m1 * N + n0] = resid[(size_t)m1 * N + n0] + v10;
      Cf[(size_t)m1 * N + n1] = resid[(size_t)m1 * N + n1] + v11;
    } else if (MODE == 2) {
      Cb[(size_t)m0 * N + n0] = f2bf(v00); Cb[(size_t)m0 * N + n1] = f2bf(v01);
      Cb[(size_t)m1 * N + n0] = f2bf(v10); Cb[(size_t)m1 * N + n1] = f2bf(v11);
    } else {
      const float c = 0.70710678118654752f;
      Cb[(size_t)m0 * N + n0] = f2bf(0.5f * v00 * (1.0f + erff(v00 * c)));
      Cb[(size_t)m0 * N + n1] = f2bf(0.5f * v01 * (1.0f + erff(v01 * c)));
      Cb[(size_t)m1 * N + n0] = f2bf(0.5f * v10 * (1.0f + erff(v10 * c)));
      Cb[(size_t)m1 * N + n1] = f2bf(0.5f * v11 * (1.0f + erff(v11 * c)));
    }
  }
}

// ---------------------------------------------------------------------------
// Split fused QKV [rows,1536] into per-head Q/K/V [B,H,T,DHEAD] (bf16)
// ---------------------------------------------------------------------------
__global__ __launch_bounds__(256)
void qkv_split_kernel(const bf16_t* __restrict__ qkv, bf16_t* __restrict__ q,
                      bf16_t* __restrict__ k, bf16_t* __restrict__ v) {
  size_t idx = (size_t)blockIdx.x * blockDim.x + threadIdx.x;
  int row   = (int)(idx / (3 * DMODEL));
  int e     = (int)(idx % (3 * DMODEL));
  int which = e / DMODEL;
  int rem   = e % DMODEL;
  int hh    = rem / DHEAD;
  int d     = rem % DHEAD;
  int bb    = row / SEQ;
  int t     = row % SEQ;
  size_t o  = (((size_t)bb * NHEADS + hh) * SEQ + t) * DHEAD + d;
  bf16_t val = qkv[idx];
  if (which == 0) q[o] = val;
  else if (which == 1) k[o] = val;
  else v[o] = val;
}

// ---------------------------------------------------------------------------
// Flash attention: grid (SEQ/64, B*H), block 128 (4 waves).
// Q (64x64) + streamed K/V (32x64) tiles all staged by TDM; K/V double-
// buffered against TENSORcnt exactly like the GEMM. Each wave owns 16 query
// rows: S = Q K^T (WMMA), online softmax (shfl_xor row reductions),
// P -> bf16 via per-wave LDS scratch, O += P V (WMMA).
// ---------------------------------------------------------------------------
__global__ __launch_bounds__(128)
void attn_kernel(const bf16_t* __restrict__ qg, const bf16_t* __restrict__ kg,
                 const bf16_t* __restrict__ vg, bf16_t* __restrict__ out) {
  __shared__ bf16_t sQ[64][64];
  __shared__ bf16_t sK[2][32][64];
  __shared__ bf16_t sV[2][32][64];
  __shared__ bf16_t sP[4][16][32];

  const int bh  = blockIdx.y;          // b*NHEADS + h
  const int q0  = blockIdx.x * 64;
  const int tid = threadIdx.x;
  const int wave = tid >> 5;
  const int lane = tid & 31;
  const int hi   = lane >> 4;
  const int ln   = lane & 15;

  const bf16_t* Q = qg + (size_t)bh * SEQ * DHEAD;
  const bf16_t* K = kg + (size_t)bh * SEQ * DHEAD;
  const bf16_t* V = vg + (size_t)bh * SEQ * DHEAD;

  const int nsteps = (q0 + 64) >> 5;   // causal: key tiles 0 .. q0+63

  if (wave == 0) {
    tdm_load_tile(lds_off(&sQ[0][0]),    Q + (size_t)q0 * DHEAD,
                  DHEAD, SEQ, DHEAD, 64);
    tdm_load_tile(lds_off(&sK[0][0][0]), K, DHEAD, SEQ, DHEAD, 32);
    tdm_load_tile(lds_off(&sV[0][0][0]), V, DHEAD, SEQ, DHEAD, 32);
    __builtin_amdgcn_s_wait_tensorcnt(2);   // in-order => Q landed
  }
  __syncthreads();

  v16bf aq[2];
  const int qrow = wave * 16 + ln;
#pragma unroll
  for (int kk = 0; kk < 2; kk++) {
#pragma unroll
    for (int i = 0; i < 8; i++) {
      aq[kk][i]     = sQ[qrow][kk * 32 + hi * 8 + i];
      aq[kk][8 + i] = sQ[qrow][kk * 32 + hi * 8 + 16 + i];
    }
  }

  v8f o_acc[4] = {{}, {}, {}, {}};
  float m_run[8], l_run[8];
#pragma unroll
  for (int j = 0; j < 8; j++) { m_run[j] = -INFINITY; l_run[j] = 0.f; }

  for (int s = 0; s < nsteps; s++) {
    const int j0 = s * 32;
    if (wave == 0) {
      if (s + 1 < nsteps) {
        int nb = (s + 1) & 1;
        tdm_load_tile(lds_off(&sK[nb][0][0]), K + (size_t)(j0 + 32) * DHEAD,
                      DHEAD, SEQ, DHEAD, 32);
        tdm_load_tile(lds_off(&sV[nb][0][0]), V + (size_t)(j0 + 32) * DHEAD,
                      DHEAD, SEQ, DHEAD, 32);
        __builtin_amdgcn_s_wait_tensorcnt(2);   // step-s K/V complete
      } else {
        __builtin_amdgcn_s_wait_tensorcnt(0);
      }
    }
    __syncthreads();

    const int cur = s & 1;

    v8f s0 = {}, s1 = {};
#pragma unroll
    for (int kk = 0; kk < 2; kk++) {
      v16bf b0, b1;
#pragma unroll
      for (int i = 0; i < 16; i++) {
        b0[i] = sK[cur][ln][kk * 32 + (i & 8) * 2 + hi * 8 + (i & 7)];
        b1[i] = sK[cur][16 + ln][kk * 32 + (i & 8) * 2 + hi * 8 + (i & 7)];
      }
      s0 = wmma_bf16(aq[kk], b0, s0);
      s1 = wmma_bf16(aq[kk], b1, s1);
    }

#pragma unroll
    for (int j = 0; j < 8; j++) {
      int m  = wave * 16 + j + 8 * hi;
      int qi = q0 + m;
      int k0i = j0 + ln, k1i = j0 + 16 + ln;
      float a = s0[j] * 0.125f;          // 1/sqrt(64)
      float b = s1[j] * 0.125f;
      float sv0 = (k0i > qi) ? -INFINITY : a;
      float sv1 = (k1i > qi) ? -INFINITY : b;

      float rmax = fmaxf(sv0, sv1);
#pragma unroll
      for (int mm = 8; mm >= 1; mm >>= 1) rmax = fmaxf(rmax, __shfl_xor(rmax, mm));
      float mnew  = fmaxf(m_run[j], rmax);
      float scale = __expf(m_run[j] - mnew);
      float p0 = __expf(sv0 - mnew);
      float p1 = __expf(sv1 - mnew);
      float rsum = p0 + p1;
#pragma unroll
      for (int mm = 8; mm >= 1; mm >>= 1) rsum += __shfl_xor(rsum, mm);
      l_run[j] = l_run[j] * scale + rsum;
      m_run[j] = mnew;
#pragma unroll
      for (int nt = 0; nt < 4; nt++) o_acc[nt][j] = o_acc[nt][j] * scale;

      sP[wave][j + 8 * hi][ln]      = f2bf(p0);   // C-layout -> A-layout
      sP[wave][j + 8 * hi][16 + ln] = f2bf(p1);
    }
    // per-wave sP region; same-wave LDS ops are in order => no barrier

    v16bf pa;
#pragma unroll
    for (int i = 0; i < 8; i++) {
      pa[i]     = sP[wave][ln][hi * 8 + i];
      pa[8 + i] = sP[wave][ln][hi * 8 + 16 + i];
    }
#pragma unroll
    for (int nt = 0; nt < 4; nt++) {
      v16bf bv;
#pragma unroll
      for (int i = 0; i < 16; i++)
        bv[i] = sV[cur][(i & 8) * 2 + hi * 8 + (i & 7)][nt * 16 + ln];
      o_acc[nt] = wmma_bf16(pa, bv, o_acc[nt]);
    }
    __syncthreads();
  }

  const int bb = bh / NHEADS, hh = bh % NHEADS;
#pragma unroll
  for (int nt = 0; nt < 4; nt++) {
#pragma unroll
    for (int j = 0; j < 8; j++) {
      int m = wave * 16 + j + 8 * hi;
      int t = q0 + m;
      float val = o_acc[nt][j] / l_run[j];
      out[((size_t)bb * SEQ + t) * DMODEL + hh * DHEAD + nt * 16 + ln] = f2bf(val);
    }
  }
}

// ---------------------------------------------------------------------------
// Host-side orchestration
// ---------------------------------------------------------------------------
extern "C" void kernel_launch(void* const* d_in, const int* in_sizes, int n_in,
                              void* d_out, int out_size, void* d_ws, size_t ws_size,
                              hipStream_t stream) {
  const int*   x       = (const int*)  d_in[0];
  const float* tok_emb = (const float*)d_in[1];
  const float* pos_emb = (const float*)d_in[2];
  const float* qkv_w   = (const float*)d_in[3];
  const float* proj_w  = (const float*)d_in[4];
  const float* ln1_w   = (const float*)d_in[5];
  const float* ln1_b   = (const float*)d_in[6];
  const float* ln2_w   = (const float*)d_in[7];
  const float* ln2_b   = (const float*)d_in[8];
  const float* fc1_w   = (const float*)d_in[9];
  const float* fc2_w   = (const float*)d_in[10];
  const float* lnf_w   = (const float*)d_in[11];
  const float* lnf_b   = (const float*)d_in[12];
  const float* head_w  = (const float*)d_in[13];

  char* ws = (char*)d_ws;
  size_t off = 0;
  auto alloc = [&](size_t bytes) -> char* {
    char* p = ws + off;
    off += (bytes + 255) & ~(size_t)255;
    return p;
  };

  float*  h      = (float*) alloc((size_t)NROWS * DMODEL * 4);
  bf16_t* xn     = (bf16_t*)alloc((size_t)NROWS * DMODEL * 2);
  bf16_t* qkvb   = (bf16_t*)alloc((size_t)NROWS * 3 * DMODEL * 2);
  bf16_t* qh     = (bf16_t*)alloc((size_t)BATCH * NHEADS * SEQ * DHEAD * 2);
  bf16_t* kh     = (bf16_t*)alloc((size_t)BATCH * NHEADS * SEQ * DHEAD * 2);
  bf16_t* vh     = (bf16_t*)alloc((size_t)BATCH * NHEADS * SEQ * DHEAD * 2);
  bf16_t* attno  = (bf16_t*)alloc((size_t)NROWS * DMODEL * 2);
  bf16_t* ffmid  = (bf16_t*)alloc((size_t)NROWS * DFF * 2);
  bf16_t* wqkv_b = (bf16_t*)alloc((size_t)NLAYERS * 3 * DMODEL * DMODEL * 2);
  bf16_t* wproj_b= (bf16_t*)alloc((size_t)NLAYERS * DMODEL * DMODEL * 2);
  bf16_t* wfc1_b = (bf16_t*)alloc((size_t)NLAYERS * DFF * DMODEL * 2);
  bf16_t* wfc2_b = (bf16_t*)alloc((size_t)NLAYERS * DMODEL * DFF * 2);
  bf16_t* whead_b= (bf16_t*)alloc((size_t)NVOCAB * DMODEL * 2);
  (void)ws_size; (void)n_in; (void)in_sizes; (void)out_size;

  auto cvt = [&](const float* s, bf16_t* d, size_t n) {
    cvt_f32_bf16_kernel<<<2048, 256, 0, stream>>>(s, d, n);
  };
  cvt(qkv_w,  wqkv_b,  (size_t)NLAYERS * 3 * DMODEL * DMODEL);
  cvt(proj_w, wproj_b, (size_t)NLAYERS * DMODEL * DMODEL);
  cvt(fc1_w,  wfc1_b,  (size_t)NLAYERS * DFF * DMODEL);
  cvt(fc2_w,  wfc2_b,  (size_t)NLAYERS * DMODEL * DFF);
  cvt(head_w, whead_b, (size_t)NVOCAB * DMODEL);

  embed_kernel<<<NROWS, 128, 0, stream>>>(x, tok_emb, pos_emb, h);

  const size_t smem_bytes =
      (2 * GEMM_A_ELEMS + 2 * GEMM_W_ELEMS) * sizeof(bf16_t); // 24 KB

  for (int l = 0; l < NLAYERS; l++) {
    const bf16_t* wq = wqkv_b  + (size_t)l * 3 * DMODEL * DMODEL;
    const bf16_t* wp = wproj_b + (size_t)l * DMODEL * DMODEL;
    const bf16_t* w1 = wfc1_b  + (size_t)l * DFF * DMODEL;
    const bf16_t* w2 = wfc2_b  + (size_t)l * DMODEL * DFF;

    layernorm_kernel<<<NROWS, 128, 0, stream>>>(h, ln1_w + l * DMODEL,
                                                ln1_b + l * DMODEL, xn);
    gemm_bf16_kernel<2><<<dim3((3 * DMODEL) / 64, NROWS / 128), 256, smem_bytes,
                          stream>>>(xn, wq, nullptr, qkvb, nullptr,
                                    NROWS, 3 * DMODEL, DMODEL);
    qkv_split_kernel<<<(NROWS * 3 * DMODEL) / 256, 256, 0, stream>>>(
        qkvb, qh, kh, vh);
    attn_kernel<<<dim3(SEQ / 64, BATCH * NHEADS), 128, 0, stream>>>(
        qh, kh, vh, attno);
    gemm_bf16_kernel<1><<<dim3(DMODEL / 64, NROWS / 128), 256, smem_bytes,
                          stream>>>(attno, wp, h, nullptr, h,
                                    NROWS, DMODEL, DMODEL);

    layernorm_kernel<<<NROWS, 128, 0, stream>>>(h, ln2_w + l * DMODEL,
                                                ln2_b + l * DMODEL, xn);
    gemm_bf16_kernel<3><<<dim3(DFF / 64, NROWS / 128), 256, smem_bytes,
                          stream>>>(xn, w1, nullptr, ffmid, nullptr,
                                    NROWS, DFF, DMODEL);
    gemm_bf16_kernel<1><<<dim3(DMODEL / 64, NROWS / 128), 256, smem_bytes,
                          stream>>>(ffmid, w2, h, nullptr, h,
                                    NROWS, DMODEL, DFF);
  }

  layernorm_kernel<<<NROWS, 128, 0, stream>>>(h, lnf_w, lnf_b, xn);
  gemm_bf16_kernel<0><<<dim3(NVOCAB / 64, NROWS / 128), 256, smem_bytes,
                        stream>>>(xn, whead_b, (float*)d_out, nullptr, nullptr,
                                  NROWS, NVOCAB, DMODEL);
}
